// LocalAttention_84112639525202
// MI455X (gfx1250) — compile-verified, matrix operand order checked by reading, not run on hardware
//
#include <hip/hip_runtime.h>
#include <hip/hip_bf16.h>

// ---------------------------------------------------------------------------
// Local windowed attention for MI455X (gfx1250, wave32, WMMA).
//   B=8, N=4096, DIM=512, H=8, DH=64, WINDOW=128, BACKWARD=1 -> J=256, NW=32
// Stage 0: pre-convert x and weights to f16 (removes cvts from GEMM K-loops)
// Stage 1: q/k/v = x @ W{q,k,v}^T   (WMMA, 64x64 register-blocked wave tiles)
// Stage 2: per-window attention     (energy f32 + attn f16 + V tile in LDS)
// Stage 3: out = ao @ Wfc^T + bfc   (WMMA, 64x64 register-blocked wave tiles)
// ---------------------------------------------------------------------------

typedef __attribute__((ext_vector_type(16))) _Float16 v16h;
typedef __attribute__((ext_vector_type(8)))  _Float16 v8h;
typedef __attribute__((ext_vector_type(8)))  float    v8f;

#define NB      8
#define NSEQ    4096
#define DIMM    512
#define NH      8
#define DH      64
#define WS      128
#define NW      32          // NSEQ / WS
#define JW      256         // WS * (BACKWARD+1)
#define BH      (NB*NH)     // 64
#define NROWS   (NB*NSEQ)   // 32768
#define SCALEF  0.04419417382415922f   // 512^-0.5
#define NEGMAX  (-3.402823466e+38f)

union Frag16 { v16h f; v8h h[2]; };

// load a 16-half WMMA operand fragment: runs [0..7] and [16..23] from base
__device__ __forceinline__ v16h ldfrag(const _Float16* p) {
    Frag16 u;
    u.h[0] = *(const v8h*)p;
    u.h[1] = *(const v8h*)(p + 16);
    return u.f;
}

// ---------------- Stage 0: fp32 -> f16 convert ------------------------------
__global__ __launch_bounds__(256)
void cvt_kernel(const float* __restrict__ src, _Float16* __restrict__ dst, int n) {
    int i = (blockIdx.x * 256 + threadIdx.x) * 8;
    if (i + 8 <= n) {
        float4 f0 = *(const float4*)(src + i);
        float4 f1 = *(const float4*)(src + i + 4);
        v8h h;
        h[0] = (_Float16)f0.x; h[1] = (_Float16)f0.y;
        h[2] = (_Float16)f0.z; h[3] = (_Float16)f0.w;
        h[4] = (_Float16)f1.x; h[5] = (_Float16)f1.y;
        h[6] = (_Float16)f1.z; h[7] = (_Float16)f1.w;
        *(v8h*)(dst + i) = h;
    }
}

// ---------------- Stage 1: fused QKV projection ----------------------------
// 64x64 per wave: 4x4 WMMA tiles; fragments are pure f16 b128 loads.
__global__ __launch_bounds__(256)
void qkv_kernel(const _Float16* __restrict__ xh,
                const _Float16* __restrict__ Wqh,
                const _Float16* __restrict__ Wkh,
                const _Float16* __restrict__ Wvh,
                _Float16* __restrict__ qf,
                _Float16* __restrict__ kf,
                _Float16* __restrict__ vf) {
    int gw   = blockIdx.x * 8 + (threadIdx.x >> 5);
    int proj = gw >> 12;              // 4096 wave tiles per projection
    int rem  = gw & 4095;
    int mt   = rem >> 3;              // 512 row-tiles of 64
    int nt   = rem & 7;               // 8 col-tiles of 64
    int lane = threadIdx.x & 31;
    int lr   = lane & 15;
    int hs   = lane >> 4;

    const _Float16* W = (proj == 0) ? Wqh : (proj == 1) ? Wkh : Wvh;
    _Float16*     dst = (proj == 0) ? qf  : (proj == 1) ? kf  : vf;

    const _Float16* ap[4];
    const _Float16* bp[4];
#pragma unroll
    for (int i = 0; i < 4; ++i) {
        ap[i] = xh + (size_t)(mt * 64 + i * 16 + lr) * DIMM;
        bp[i] = W  + (size_t)(nt * 64 + i * 16 + lr) * DIMM;
    }

    v8f c[4][4];
#pragma unroll
    for (int i = 0; i < 4; ++i)
#pragma unroll
        for (int j = 0; j < 4; ++j) c[i][j] = (v8f){};

    for (int ks = 0; ks < DIMM; ks += 32) {
        int kb = ks + hs * 8;
        v16h a[4], bfr[4];
#pragma unroll
        for (int t = 0; t < 4; ++t) {
            a[t]   = ldfrag(ap[t] + kb);
            bfr[t] = ldfrag(bp[t] + kb);
        }
#pragma unroll
        for (int mi = 0; mi < 4; ++mi)
#pragma unroll
            for (int ni = 0; ni < 4; ++ni)
                c[mi][ni] = __builtin_amdgcn_wmma_f32_16x16x32_f16(
                    false, a[mi], false, bfr[ni], (short)0, c[mi][ni], false, false);
    }

    // store folded (b*h, n, d) in f16
#pragma unroll
    for (int mi = 0; mi < 4; ++mi) {
#pragma unroll
        for (int ni = 0; ni < 4; ++ni) {
#pragma unroll
            for (int r = 0; r < 8; ++r) {
                int row = mt * 64 + mi * 16 + r + 8 * hs;
                int b   = row >> 12;
                int n   = row & 4095;
                int col = nt * 64 + ni * 16 + lr;
                int h   = col >> 6;
                int dd  = col & 63;
                dst[(((size_t)(b * NH + h) * NSEQ) + n) * DH + dd] = (_Float16)c[mi][ni][r];
            }
        }
    }
}

// ---------------- Stage 2: windowed attention -------------------------------
// one block per (bh, window, half-of-128-rows); 8 waves.
// LDS: 64KB f32 energy + 32KB f16 attn + 32KB f16 V tile = 128KB (<320KB WGP)
__global__ __launch_bounds__(256)
void attn_kernel(const _Float16* __restrict__ qf,
                 const _Float16* __restrict__ kf,
                 const _Float16* __restrict__ vf,
                 const unsigned char* __restrict__ maskp,
                 float* __restrict__ attn_out,
                 _Float16* __restrict__ aof) {
    __shared__ float    sE[64 * JW];
    __shared__ _Float16 sAh[64 * JW];
    __shared__ _Float16 sV[JW * DH];

    int blk  = blockIdx.x;
    int half = blk & 1;
    int w    = (blk >> 1) & (NW - 1);
    int bh   = blk >> 6;
    int b    = bh >> 3;
    int tid  = threadIdx.x;
    int wave = tid >> 5;
    int lane = tid & 31;
    int lr   = lane & 15;
    int hs   = lane >> 4;
    int row0 = w * WS + half * 64;                  // first query position

    // ---- stage V look-around tile into LDS (pad rows = -1.0) --------------
    {
        int krow = tid;                             // key row 0..255
        int p    = (w - 1) * WS + krow;
        uint4* dstv = (uint4*)(sV + krow * DH);
        if (p >= 0) {
            const uint4* srcv = (const uint4*)(vf + ((size_t)bh * NSEQ + p) * DH);
#pragma unroll
            for (int i = 0; i < 8; ++i) dstv[i] = srcv[i];
        } else {
            uint4 fill;
            fill.x = fill.y = fill.z = fill.w = 0xBC00BC00u;   // f16 -1.0 pairs
#pragma unroll
            for (int i = 0; i < 8; ++i) dstv[i] = fill;
        }
    }

    const _Float16* qbase = qf + ((size_t)bh * NSEQ + row0) * DH;

    // ---- energy = q @ k^T : hoist q fragments, 8 key tiles per wave --------
    {
        int mt  = wave >> 1;                        // 0..3 (query tile)
        int nt0 = (wave & 1) * 8;                   // key tile base
        const _Float16* ap = qbase + (size_t)(mt * 16 + lr) * DH;
        int kb0 = hs * 8;
        v16h a0 = ldfrag(ap + kb0);
        v16h a1 = ldfrag(ap + kb0 + 32);
        v16h negone;
#pragma unroll
        for (int i = 0; i < 16; ++i) negone[i] = (_Float16)(-1.0f);
#pragma unroll
        for (int ni = 0; ni < 8; ++ni) {
            int nt = nt0 + ni;
            int j  = nt * 16 + lr;
            int p  = (w - 1) * WS + j;
            v16h b0, b1;
            if (p >= 0) {
                const _Float16* bp = kf + ((size_t)bh * NSEQ + p) * DH;
                b0 = ldfrag(bp + kb0);
                b1 = ldfrag(bp + kb0 + 32);
            } else {
                b0 = negone;
                b1 = negone;
            }
            v8f c = {};
            c = __builtin_amdgcn_wmma_f32_16x16x32_f16(false, a0, false, b0,
                                                       (short)0, c, false, false);
            c = __builtin_amdgcn_wmma_f32_16x16x32_f16(false, a1, false, b1,
                                                       (short)0, c, false, false);
#pragma unroll
            for (int r = 0; r < 8; ++r)
                sE[(mt * 16 + r + 8 * hs) * JW + j] = c[r] * SCALEF;
        }
    }
    __syncthreads();

    // ---- masked softmax: 4 lanes per row, shfl_xor reductions --------------
    {
        int row  = tid >> 2;                        // 0..63
        int part = tid & 3;
        int qpos = row0 + row;
        bool mq  = maskp[(size_t)b * NSEQ + qpos] != 0;
        float* er = sE + row * JW;
        int j0 = part * 64;
        float mmax = NEGMAX;
        for (int j = j0; j < j0 + 64; ++j) {
            int p = (w - 1) * WS + j;
            bool valid = (p >= 0) && (p <= qpos) && mq &&
                         (maskp[(size_t)b * NSEQ + (p < 0 ? 0 : p)] != 0);
            float e = valid ? er[j] : NEGMAX;
            er[j] = e;
            mmax  = fmaxf(mmax, e);
        }
        mmax = fmaxf(mmax, __shfl_xor(mmax, 1, 32));
        mmax = fmaxf(mmax, __shfl_xor(mmax, 2, 32));
        float s = 0.0f;
        for (int j = j0; j < j0 + 64; ++j) {
            float e = __expf(er[j] - mmax);
            er[j] = e;
            s += e;
        }
        s += __shfl_xor(s, 1, 32);
        s += __shfl_xor(s, 2, 32);
        float inv = 1.0f / s;
        float* aptr = attn_out +
            (((size_t)bh * NW + w) * WS + half * 64 + row) * JW;
        _Float16* ah = sAh + row * JW;
        for (int j = j0; j < j0 + 64; ++j) {
            float a = er[j] * inv;
            ah[j]   = (_Float16)a;                  // f16 attn for the AV WMMA
            aptr[j] = a;                            // second reference output
        }
    }
    __syncthreads();

    // ---- out = attn @ v : A from sAh (b128 LDS), B from sV ----------------
    {
        int mt  = wave >> 1;                        // 0..3
        int nt0 = (wave & 1) * 2;                   // 0 or 2
        v8f c2[2];
        c2[0] = (v8f){};
        c2[1] = (v8f){};
        const _Float16* arow = sAh + (mt * 16 + lr) * JW;
        for (int ks = 0; ks < JW; ks += 32) {
            int kb = ks + hs * 8;
            v16h a = ldfrag(arow + kb);
#pragma unroll
            for (int nn = 0; nn < 2; ++nn) {
                int n = (nt0 + nn) * 16 + lr;       // head-dim column
                v16h bfr;
#pragma unroll
                for (int i = 0; i < 8; ++i) {
                    bfr[i]     = sV[(kb + i) * DH + n];
                    bfr[8 + i] = sV[(kb + 16 + i) * DH + n];
                }
                c2[nn] = __builtin_amdgcn_wmma_f32_16x16x32_f16(
                    false, a, false, bfr, (short)0, c2[nn], false, false);
            }
        }
#pragma unroll
        for (int nn = 0; nn < 2; ++nn) {
#pragma unroll
            for (int r = 0; r < 8; ++r) {
                int mi = mt * 16 + r + 8 * hs;
                aof[((size_t)bh * NSEQ + row0 + mi) * DH + (nt0 + nn) * 16 + lr] =
                    (_Float16)c2[nn][r];
            }
        }
    }
}

// ---------------- Stage 3: final FC  out = ao @ Wfc^T + bfc -----------------
// 64x64 per wave; (32768/64)*(512/64) = 4096 waves -> 512 blocks
__global__ __launch_bounds__(256)
void fc_kernel(const _Float16* __restrict__ aof,
               const _Float16* __restrict__ Wfch,
               const float* __restrict__ bfc,
               float* __restrict__ outp) {
    int gw   = blockIdx.x * 8 + (threadIdx.x >> 5);
    int mt   = gw >> 3;
    int nt   = gw & 7;
    int lane = threadIdx.x & 31;
    int lr   = lane & 15;
    int hs   = lane >> 4;

    int arow[4];
    const _Float16* bp[4];
#pragma unroll
    for (int i = 0; i < 4; ++i) {
        arow[i] = mt * 64 + i * 16 + lr;
        bp[i]   = Wfch + (size_t)(nt * 64 + i * 16 + lr) * DIMM;
    }

    v8f c[4][4];
#pragma unroll
    for (int i = 0; i < 4; ++i)
#pragma unroll
        for (int j = 0; j < 4; ++j) c[i][j] = (v8f){};

    for (int ks = 0; ks < DIMM; ks += 32) {
        int kb = ks + hs * 8;
        v16h a[4], bfr[4];
#pragma unroll
        for (int t = 0; t < 4; ++t) {
            int b = arow[t] >> 12;
            int n = arow[t] & 4095;
            // unfold (h, dd): each 8-half run stays inside one head slice
            int k0 = kb, k1 = kb + 16;
            const _Float16* a0 =
                aof + (((size_t)(b * NH + (k0 >> 6)) * NSEQ) + n) * DH + (k0 & 63);
            const _Float16* a1 =
                aof + (((size_t)(b * NH + (k1 >> 6)) * NSEQ) + n) * DH + (k1 & 63);
            Frag16 u;
            u.h[0] = *(const v8h*)a0;
            u.h[1] = *(const v8h*)a1;
            a[t]   = u.f;
            bfr[t] = ldfrag(bp[t] + kb);
        }
#pragma unroll
        for (int mi = 0; mi < 4; ++mi)
#pragma unroll
            for (int ni = 0; ni < 4; ++ni)
                c[mi][ni] = __builtin_amdgcn_wmma_f32_16x16x32_f16(
                    false, a[mi], false, bfr[ni], (short)0, c[mi][ni], false, false);
    }

#pragma unroll
    for (int mi = 0; mi < 4; ++mi) {
#pragma unroll
        for (int ni = 0; ni < 4; ++ni) {
#pragma unroll
            for (int r = 0; r < 8; ++r) {
                int row = mt * 64 + mi * 16 + r + 8 * hs;
                int col = nt * 64 + ni * 16 + lr;
                outp[(size_t)row * DIMM + col] = c[mi][ni][r] + bfc[col];
            }
        }
    }
}

// ---------------------------------------------------------------------------
extern "C" void kernel_launch(void* const* d_in, const int* in_sizes, int n_in,
                              void* d_out, int out_size, void* d_ws, size_t ws_size,
                              hipStream_t stream) {
    const float* x   = (const float*)d_in[0];
    const float* Wq  = (const float*)d_in[1];
    const float* Wk  = (const float*)d_in[2];
    const float* Wv  = (const float*)d_in[3];
    const float* Wfc = (const float*)d_in[4];
    const float* bfc = (const float*)d_in[5];
    const unsigned char* mask = (const unsigned char*)d_in[6];

    float* outp  = (float*)d_out;
    float* attnp = outp + (size_t)NB * NSEQ * DIMM;     // second output, flat

    // f16 workspace: q, k, v, ao, xh (each 32MB) + 4 weight copies (0.5MB ea)
    const size_t elems = (size_t)BH * NSEQ * DH;        // 16,777,216
    const size_t welem = (size_t)DIMM * DIMM;           // 262,144
    _Float16* qf   = (_Float16*)d_ws;
    _Float16* kf   = qf + elems;
    _Float16* vf   = kf + elems;
    _Float16* aof  = vf + elems;
    _Float16* xh   = aof + elems;
    _Float16* Wqh  = xh + elems;
    _Float16* Wkh  = Wqh + welem;
    _Float16* Wvh  = Wkh + welem;
    _Float16* Wfch = Wvh + welem;

    // Stage 0: one-time f16 conversion (x: 16M elems, weights: 256K each)
    cvt_kernel<<<8192, 256, 0, stream>>>(x,   xh,   (int)elems);
    cvt_kernel<<<128,  256, 0, stream>>>(Wq,  Wqh,  (int)welem);
    cvt_kernel<<<128,  256, 0, stream>>>(Wk,  Wkh,  (int)welem);
    cvt_kernel<<<128,  256, 0, stream>>>(Wv,  Wvh,  (int)welem);
    cvt_kernel<<<128,  256, 0, stream>>>(Wfc, Wfch, (int)welem);

    qkv_kernel<<<1536, 256, 0, stream>>>(xh, Wqh, Wkh, Wvh, qf, kf, vf);
    attn_kernel<<<4096, 256, 0, stream>>>(qf, kf, vf, mask, attnp, aof);
    fc_kernel<<<512, 256, 0, stream>>>(aof, Wfch, bfc, outp);
}
